// MyModel_60619168416012
// MI455X (gfx1250) — matrix operand order repaired and checked by reading, not compiled, not run on hardware
//
#include <hip/hip_runtime.h>
#include <hip/hip_bf16.h>

// ---------------------------------------------------------------------------
// Problem dims (compile-time constants from the reference)
// ---------------------------------------------------------------------------
#define Bn 64
#define Ln 256
#define En 128
#define Cn 16
#define Kn 5
#define NLn 4
#define STEPF 0.1f
#define LDST 136              // LDS row stride (bf16 elems): 272B -> 4-bank skew

typedef __attribute__((ext_vector_type(16))) __bf16 v16bf;
typedef __attribute__((ext_vector_type(8)))  __bf16 v8bf;
typedef __attribute__((ext_vector_type(8)))  float  v8f;

// ---------------------------------------------------------------------------
// bf16 <-> f32 helpers (bit ops only)
// ---------------------------------------------------------------------------
__device__ __forceinline__ __bf16 f2bf(float f) {
    union { float f; unsigned u; } v; v.f = f;
    unsigned r = v.u + 0x7FFFu + ((v.u >> 16) & 1u);
    union { unsigned short s; __bf16 b; } o; o.s = (unsigned short)(r >> 16);
    return o.b;
}
__device__ __forceinline__ float bf2f(__bf16 b) {
    union { unsigned short s; __bf16 b; } i; i.b = b;
    union { unsigned u; float f; } o; o.u = ((unsigned)i.s) << 16;
    return o.f;
}

// Native CDNA5 transcendental tanh (TRANS op) -- replaces libm's ~40-op
// branchy expansion with a single VALU instruction.
__device__ __forceinline__ float fast_tanh(float x) {
    float r;
    asm("v_tanh_f32 %0, %1" : "=v"(r) : "v"(x));
    return r;
}

// Async global->LDS copy (16B per lane), tracked by ASYNCcnt (ISA 10.x/15.18).
__device__ __forceinline__ void async_g2l_b128(unsigned lds_off,
                                               unsigned long long gaddr) {
    asm volatile("global_load_async_to_lds_b128 %0, %1, off"
                 :: "v"(lds_off), "v"(gaddr) : "memory");
}
__device__ __forceinline__ void wait_async0() {
    asm volatile("s_wait_asynccnt 0" ::: "memory");
}

// D = A(16x32 bf16) x B(32x16 bf16) + C(16x16 f32), wave32 WMMA
__device__ __forceinline__ v8f wmma_bf16(v16bf a, v16bf b, v8f c) {
    return __builtin_amdgcn_wmma_f32_16x16x32_bf16(
        false, a, false, b, (short)0, c, false, false);
}

// Fragment layouts used here (wave32; D as in ISA 7.12.2):
//  A (16x32, 16-bit): lane%16 = M; k order used here (not ISA table order): half*16 + 0..15
//  B (32x16, 16-bit): lane%16 = N; k order used here (not ISA table order): half*16 + 0..15
//  D (16x16 f32): VGPR r -> M = r + 8*half, N = lane%16
__device__ __forceinline__ v16bf ldB16(const __bf16* p) {   // 16 contiguous
    union { v16bf v; v8bf h[2]; } u;
    u.h[0] = *(const v8bf*)p;
    u.h[1] = *(const v8bf*)(p + 8);
    return u.v;
}
__device__ __forceinline__ v16bf ldA16(const __bf16* p, int half) {
    union { v16bf v; v8bf h[2]; } u;
    u.h[0] = *(const v8bf*)(p + half * 16);
    u.h[1] = *(const v8bf*)(p + half * 16 + 8);
    return u.v;
}

// ===========================================================================
// Repack kernels (one-shot, tiny): f32 -> bf16, conv weights to tap-major.
// ===========================================================================
__global__ void k_repack_w(const float* __restrict__ cw, __bf16* __restrict__ wp) {
    const int o = blockIdx.x, layer = blockIdx.y;
    for (int kk = threadIdx.x; kk < En * Kn; kk += 256) {
        int tap = kk >> 7, e = kk & (En - 1);
        wp[((size_t)layer * En + o) * (En * Kn) + kk] =
            f2bf(cw[(((size_t)layer * En + o) * En + e) * Kn + tap]);
    }
}
__global__ void k_repack_t1(const float* __restrict__ w, __bf16* __restrict__ wp) {
    size_t idx = (size_t)blockIdx.x * 256 + threadIdx.x;
    wp[idx] = f2bf(w[idx]);
}

// ===========================================================================
// Stage 1: embedding + 4 residual conv layers (native v_tanh_f32).
// h TRANSPOSED in LDS: hT[l+2][e], stride LDST, 2-row zero halo (SAME pad,
// branch-free). Double-buffered. N-blocked by 4 (A-fragment reuse x4);
// K-loop rolled (unroll 2) so LICM cannot hoist+spill A loads.
// ===========================================================================
__global__ void k_conv(const int* __restrict__ aa,
                       const float* __restrict__ emb,
                       const __bf16* __restrict__ wp,
                       const float* __restrict__ conv_b,
                       __bf16* __restrict__ x_out) {
    extern __shared__ __bf16 hsm[];
    __bf16* h0 = hsm;                       // (Ln+4) x LDST
    __bf16* h1 = hsm + (Ln + 4) * LDST;

    const int b    = blockIdx.x;
    const int tid  = threadIdx.x;
    const int lane = tid & 31;
    const int wave = tid >> 5;
    const int half = lane >> 4;
    const int mrow = lane & 15;

    // zero halo rows (l = -2,-1,256,257) of both buffers
    for (int idx = tid; idx < 4 * LDST; idx += 256) {
        int rr = idx / LDST, cc = idx - rr * LDST;
        int row = (rr < 2) ? rr : (Ln + rr);
        h0[row * LDST + cc] = f2bf(0.0f);
        h1[row * LDST + cc] = f2bf(0.0f);
    }
    // embedding gather: one row per thread, 16B LDS stores
    {
        int tok = aa[b * Ln + tid];
        const float* er = emb + (size_t)tok * En;
        __bf16* dr = h0 + (tid + 2) * LDST;
#pragma unroll
        for (int i = 0; i < En; i += 8) {
            v8bf t;
#pragma unroll
            for (int r = 0; r < 8; ++r) t[r] = f2bf(er[i + r]);
            *(v8bf*)(dr + i) = t;
        }
    }
    __syncthreads();

    const int o0    = wave * 16;
    const int obase = o0 + 8 * half;

    for (int layer = 0; layer < NLn; ++layer) {
        const __bf16* hcur = (layer & 1) ? h1 : h0;
        __bf16*       hnxt = (layer & 1) ? h0 : h1;
        const __bf16* wrow = wp + ((size_t)layer * En + o0 + mrow) * (En * Kn);

        float bias[8];
#pragma unroll
        for (int r = 0; r < 8; ++r) bias[r] = conv_b[layer * En + obase + r];

        for (int ntb = 0; ntb < Ln / 16; ntb += 4) {
            const int lbase = ntb * 16;
            v8f acc0 = {}, acc1 = {}, acc2 = {}, acc3 = {};
#pragma clang loop unroll_count(2)
            for (int ks = 0; ks < (En * Kn) / 32; ++ks) {
                const int tap = ks >> 2;
                const int es  = (ks & 3) * 32;
                v16bf afr = ldA16(wrow + tap * En + es, half);
                const __bf16* hb =
                    hcur + (lbase + mrow + tap) * LDST + es + half * 16;
                v16bf b0 = ldB16(hb);
                v16bf b1 = ldB16(hb + 16 * LDST);
                v16bf b2 = ldB16(hb + 32 * LDST);
                v16bf b3 = ldB16(hb + 48 * LDST);
                acc0 = wmma_bf16(afr, b0, acc0);
                acc1 = wmma_bf16(afr, b1, acc1);
                acc2 = wmma_bf16(afr, b2, acc2);
                acc3 = wmma_bf16(afr, b3, acc3);
            }
            // epilogue: h_next = tanh(h + y + bias), native v_tanh_f32
#pragma unroll
            for (int q = 0; q < 4; ++q) {
                const v8f accq = (q == 0) ? acc0 : (q == 1) ? acc1
                               : (q == 2) ? acc2 : acc3;
                const int l = lbase + q * 16 + mrow;
                v8bf hvec = *(const v8bf*)(hcur + (l + 2) * LDST + obase);
                v8bf outv;
#pragma unroll
                for (int r = 0; r < 8; ++r)
                    outv[r] = f2bf(fast_tanh(bf2f(hvec[r]) + accq[r] + bias[r]));
                *(v8bf*)(hnxt + (l + 2) * LDST + obase) = outv;
            }
        }
        __syncthreads();
    }

    // final (NL even -> h0); x[b][l][e] row-major per l
    {
        const uint4* src = (const uint4*)(h0 + (tid + 2) * LDST);
        uint4* dst = (uint4*)(x_out + ((size_t)b * Ln + tid) * En);
#pragma unroll
        for (int i = 0; i < En / 8; ++i) dst[i] = src[i];
    }
}

// ===========================================================================
// Stage 2: x2 = x @ t1_w^T + t1_b, output layout chosen for stage 3's B:
//   x2v[b][j][c][e] = x2[b][j][e*16 + c]
// ===========================================================================
__global__ void k_t1(const __bf16* __restrict__ x,
                     const __bf16* __restrict__ t1wp,
                     const float* __restrict__ t1_b,
                     __bf16* __restrict__ x2v) {
    const int b    = blockIdx.x;
    const int lane = threadIdx.x & 31;
    const int wave = threadIdx.x >> 5;
    const int half = lane >> 4;
    const int mrow = lane & 15;

    for (int it = 0; it < 16; ++it) {
        int tile = blockIdx.y * 128 + it * 8 + wave;   // 0..2047
        int Mt = tile & 15;          // j-tile
        int Nt = tile >> 4;          // e_out
        int j0 = Mt * 16;
        int f0 = Nt * 16;

        v8f acc = {};
        const __bf16* xrow = x + ((size_t)b * Ln + j0 + mrow) * En;
        const __bf16* wrow = t1wp + (size_t)(f0 + mrow) * En;     // n = mrow
#pragma unroll
        for (int e0 = 0; e0 < En; e0 += 32) {
            v16bf afr = ldA16(xrow + e0, half);
            v16bf bfr = ldB16(wrow + e0 + half * 16);
            acc = wmma_bf16(afr, bfr, acc);
        }
        float biasv = t1_b[f0 + mrow];
#pragma unroll
        for (int r = 0; r < 8; ++r) {
            int j = j0 + r + 8 * half;
            x2v[(((size_t)b * Ln + j) * Cn + mrow) * En + Nt] =
                f2bf(acc[r] + biasv);
        }
    }
}

// ===========================================================================
// Stage 3 (dominant GEMM, fused softmax + aggregation):
//  - x[b] (64KB) staged into LDS ONCE per block via global_load_async_to_lds
//    (ASYNCcnt path); 8 waves then read A-fragments from LDS (8x less L2
//    traffic). Rows padded to LDST for conflict-free A-fragment loads.
//  - B operand (x2v[b][a][c][:]) is d-invariant: 4 fragments preloaded once.
//  - Softmax over c = shfl within 16-lane groups; fused s / sum reduction.
// ===========================================================================
__global__ void k_scores(const __bf16* __restrict__ x,
                         const __bf16* __restrict__ x2v,
                         const float* __restrict__ bb,
                         float* __restrict__ s_out,
                         float* __restrict__ ssum_out) {
    extern __shared__ __bf16 smem[];
    __bf16* xsh  = smem;                          // Ln x LDST bf16
    float*  bbsh = (float*)(smem + Ln * LDST);    // Ln*3 f32

    const int b    = blockIdx.x;
    const int tid  = threadIdx.x;
    const int lane = tid & 31;
    const int wave = tid >> 5;
    const int half = lane >> 4;
    const int mrow = lane & 15;

    // async-stage x[b] into LDS: 4096 x 16B transfers, row-padded layout
    {
        const unsigned xsh_base = (unsigned)(size_t)xsh;
        const char* gsrc = (const char*)(x + (size_t)b * Ln * En);
        for (int i = tid; i < (Ln * En) / 8; i += 256) {     // 16 iters
            int row = i >> 4;                 // 16 x 16B per 256B row
            int c16 = (i & 15) * 16;
            async_g2l_b128(xsh_base + (unsigned)(row * (LDST * 2) + c16),
                           (unsigned long long)(gsrc + i * 16));
        }
    }
    for (int i = tid; i < Ln * 3; i += 256)
        bbsh[i] = bb[(size_t)b * Ln * 3 + i];
    wait_async0();
    __syncthreads();

    const int a = blockIdx.y * 8 + wave;   // 0..255
    const float bax = bbsh[a * 3 + 0];
    const float bay = bbsh[a * 3 + 1];
    const float baz = bbsh[a * 3 + 2];

    // preload the 4 d-invariant B fragments
    const __bf16* brow = x2v + (((size_t)b * Ln + a) * Cn + mrow) * En + half * 16;
    v16bf bfr[4];
#pragma unroll
    for (int i = 0; i < 4; ++i) bfr[i] = ldB16(brow + i * 32);

    float as0 = 0.f, as1 = 0.f, as2 = 0.f, asum = 0.f;

    for (int d0 = 0; d0 < Ln; d0 += 16) {
        v8f acc = {};
        const __bf16* xrow = xsh + (d0 + mrow) * LDST;
#pragma unroll
        for (int i = 0; i < 4; ++i) {
            v16bf afr = ldA16(xrow + i * 32, half);
            acc = wmma_bf16(afr, bfr[i], acc);
        }
#pragma unroll
        for (int r = 0; r < 8; ++r) {
            float sc = acc[r];
            float mx = sc;
#pragma unroll
            for (int m = 1; m < 16; m <<= 1)
                mx = fmaxf(mx, __shfl_xor(mx, m, 32));
            float ex  = __expf(sc - mx);
            float sum = ex;
#pragma unroll
            for (int m = 1; m < 16; m <<= 1)
                sum += __shfl_xor(sum, m, 32);
            float sm = ex / sum;

            int d = d0 + r + 8 * half;
            as0  += sm * (bax - bbsh[d * 3 + 0]);
            as1  += sm * (bay - bbsh[d * 3 + 1]);
            as2  += sm * (baz - bbsh[d * 3 + 2]);
            asum += sm;
        }
    }
    as0  += __shfl_xor(as0, 16, 32);
    as1  += __shfl_xor(as1, 16, 32);
    as2  += __shfl_xor(as2, 16, 32);
    asum += __shfl_xor(asum, 16, 32);

    if (half == 0) {
        size_t base = ((size_t)b * Cn + mrow) * Ln + a;
        s_out[base * 3 + 0] = as0;
        s_out[base * 3 + 1] = as1;
        s_out[base * 3 + 2] = as2;
        ssum_out[base]      = asum;
    }
}

// ===========================================================================
// Stage 4: vd[b,a,k] = sum_{c,m} W2[k,c,m]*s[b,c,a,m] + sum_c b2[k,c]*ssum
// ===========================================================================
__global__ void k_vd(const float* __restrict__ s,
                     const float* __restrict__ ssum,
                     const float* __restrict__ bb,
                     const float* __restrict__ t2_w,
                     const float* __restrict__ t2_b,
                     float* __restrict__ newbb) {
    __shared__ float w2s[3 * Cn * 3];
    __shared__ float b2s[3 * Cn];
    const int b = blockIdx.x;
    const int a = threadIdx.x;
    if (a < 3 * Cn * 3) w2s[a] = t2_w[a];
    if (a < 3 * Cn)     b2s[a] = t2_b[a];
    __syncthreads();

    float vd[3] = {0.f, 0.f, 0.f};
    for (int c = 0; c < Cn; ++c) {
        size_t sb = ((size_t)b * Cn + c) * Ln + a;
        float s0 = s[sb * 3 + 0];
        float s1 = s[sb * 3 + 1];
        float s2 = s[sb * 3 + 2];
        float su = ssum[sb];
#pragma unroll
        for (int k = 0; k < 3; ++k) {
            const float* w = w2s + (k * Cn + c) * 3;
            vd[k] += w[0] * s0 + w[1] * s1 + w[2] * s2 + b2s[k * Cn + c] * su;
        }
    }
#pragma unroll
    for (int k = 0; k < 3; ++k) {
        size_t idx = ((size_t)b * Ln + a) * 3 + k;
        newbb[idx] = bb[idx] + STEPF * vd[k];
    }
}

// ===========================================================================
// Stage 5: mse[bi] = mean_a sum_k (new_bb[bi+1,a,k] - bb[bi,a,k])^2
// ===========================================================================
__global__ void k_mse(const float* __restrict__ newbb,
                      const float* __restrict__ bb,
                      float* __restrict__ out) {
    __shared__ float red[256];
    const int bi = blockIdx.x;
    const int a  = threadIdx.x;
    const float* nb = newbb + ((size_t)(bi + 1) * Ln + a) * 3;
    const float* ob = bb + ((size_t)bi * Ln + a) * 3;
    float d0 = nb[0] - ob[0];
    float d1 = nb[1] - ob[1];
    float d2 = nb[2] - ob[2];
    red[a] = d0 * d0 + d1 * d1 + d2 * d2;
    __syncthreads();
    for (int st = 128; st > 0; st >>= 1) {
        if (a < st) red[a] += red[a + st];
        __syncthreads();
    }
    if (a == 0) out[bi] = red[0] / (float)Ln;
}

// ===========================================================================
// Host launcher
// ===========================================================================
extern "C" void kernel_launch(void* const* d_in, const int* in_sizes, int n_in,
                              void* d_out, int out_size, void* d_ws, size_t ws_size,
                              hipStream_t stream) {
    const int*   aa     = (const int*)d_in[0];
    const float* bb     = (const float*)d_in[1];
    const float* emb    = (const float*)d_in[2];
    const float* conv_w = (const float*)d_in[3];
    const float* conv_b = (const float*)d_in[4];
    const float* t1_w   = (const float*)d_in[5];
    const float* t1_b   = (const float*)d_in[6];
    const float* t2_w   = (const float*)d_in[7];
    const float* t2_b   = (const float*)d_in[8];
    float* out = (float*)d_out;

    auto alignup = [](size_t x) { return (x + 255) & ~(size_t)255; };
    char* ws = (char*)d_ws;
    size_t off = 0;
    __bf16* ws_x    = (__bf16*)(ws + off); off += alignup((size_t)Bn * Ln * En * 2);
    __bf16* ws_x2v  = (__bf16*)(ws + off); off += alignup((size_t)Bn * Ln * Cn * En * 2);
    __bf16* ws_wp   = (__bf16*)(ws + off); off += alignup((size_t)NLn * En * En * Kn * 2);
    __bf16* ws_t1wp = (__bf16*)(ws + off); off += alignup((size_t)En * Cn * En * 2);
    float* ws_s     = (float*)(ws + off);  off += alignup((size_t)Bn * Cn * Ln * 3 * 4);
    float* ws_ssum  = (float*)(ws + off);  off += alignup((size_t)Bn * Cn * Ln * 4);
    float* ws_nbb   = (float*)(ws + off);  off += alignup((size_t)Bn * Ln * 3 * 4);
    (void)ws_size; (void)in_sizes; (void)n_in; (void)out_size;

    const size_t conv_lds   = (size_t)2 * (Ln + 4) * LDST * sizeof(__bf16); // ~141 KB
    const size_t scores_lds = (size_t)Ln * LDST * sizeof(__bf16)            // x stage
                            + (size_t)Ln * 3 * sizeof(float);               // bb

    k_repack_w <<<dim3(En, NLn),            dim3(256), 0, stream>>>(conv_w, ws_wp);
    k_repack_t1<<<dim3(En * Cn * En / 256), dim3(256), 0, stream>>>(t1_w, ws_t1wp);
    k_conv     <<<dim3(Bn),       dim3(256), conv_lds, stream>>>(aa, emb, ws_wp, conv_b, ws_x);
    k_t1       <<<dim3(Bn, 16),   dim3(256), 0, stream>>>(ws_x, ws_t1wp, t1_b, ws_x2v);
    k_scores   <<<dim3(Bn, 32),   dim3(256), scores_lds, stream>>>(ws_x, ws_x2v, bb, ws_s, ws_ssum);
    k_vd       <<<dim3(Bn),       dim3(256), 0, stream>>>(ws_s, ws_ssum, bb, t2_w, t2_b, ws_nbb);
    k_mse      <<<dim3(Bn - 1),   dim3(256), 0, stream>>>(ws_nbb, bb, out);
}
